// MultiheadAttention_10488310137284
// MI455X (gfx1250) — compile-verified, matrix operand order checked by reading, not run on hardware
//
#include <hip/hip_runtime.h>

// ---- problem constants (match reference) ----
#define BATCH   2
#define SEQ     4096
#define DIN     512
#define EMB     512
#define NHEAD   8
#define HDIM    64
#define MTOT    (BATCH * SEQ)     // 8192 rows
#define QKVN    (3 * EMB)         // 1536

typedef __bf16 bf16_t;
typedef __attribute__((ext_vector_type(16))) __bf16 v16bf;
typedef __attribute__((ext_vector_type(8)))  float  v8f;

union Frag {
    unsigned u[8];
    v16bf    v;
};

// Per-lane dword offset inside a 16x32 bf16 A/B fragment (ISA 16-bit A layout):
// VGPR v holds K = (v/4)*16 + half*8 + (v%4)*2  and  K+1   (half = lane/16)
__device__ __forceinline__ int k0of(int v, int half) {
    return ((v >> 2) << 4) + (half << 3) + ((v & 3) << 1);
}

// Load one 16x32 bf16 fragment from a K-contiguous source (global or LDS).
// row_base points at element (row_or_col = lane%16, kbase) of the matrix.
__device__ __forceinline__ Frag load_frag(const bf16_t* row_base, int half) {
    Frag f;
    const char* p = (const char*)row_base;
#pragma unroll
    for (int v = 0; v < 8; ++v) {
        f.u[v] = *(const unsigned*)(p + 2 * k0of(v, half));
    }
    return f;
}

__device__ __forceinline__ v8f wmma_bf16(const Frag& a, const Frag& b, v8f c) {
    return __builtin_amdgcn_wmma_f32_16x16x32_bf16(
        /*neg_a=*/false, a.v, /*neg_b=*/false, b.v,
        /*c_mod=*/(short)0, c, /*reuse_a=*/false, /*reuse_b=*/false);
}

// LDS byte offset of a __shared__ object (low 32 bits of the generic pointer).
__device__ __forceinline__ unsigned lds_off(const void* p) {
    return (unsigned)(unsigned long long)p;
}

// Async copy of 16 bytes per lane: global -> LDS (tracked by ASYNCcnt).
__device__ __forceinline__ void async_b128(unsigned lds_byte_off, const void* gaddr) {
    asm volatile("global_load_async_to_lds_b128 %0, %1, off"
                 :: "v"(lds_byte_off), "v"(gaddr)
                 : "memory");
}

// ---------------- stage 0: fp32 -> bf16 ----------------
__global__ void cvt_bf16_kernel(const float* __restrict__ x,
                                bf16_t* __restrict__ y, int n) {
    int i = blockIdx.x * blockDim.x + threadIdx.x;
    if (i < n) y[i] = (bf16_t)x[i];
}

// ---------------- stage 1: fused QKV projection ----------------
// C[MTOT x QKVN] = Xb[MTOT x 512] * Wqkv^T (+ bias), 32x32 tile per wave.
// scatter into Q*scale[bh][s][d], K[bh][s][d], Vt[bh][d][s]  (bf16)
__global__ __launch_bounds__(256)
void qkv_gemm_kernel(const bf16_t* __restrict__ Xb,
                     const bf16_t* __restrict__ Wb,
                     const float*  __restrict__ bias,
                     bf16_t* __restrict__ Qb,
                     bf16_t* __restrict__ Kb,
                     bf16_t* __restrict__ Vt) {
    const int wid  = (blockIdx.x * blockDim.x + threadIdx.x) >> 5;
    const int lane = threadIdx.x & 31;
    const int half = lane >> 4;
    const int lq   = lane & 15;

    const int tilesN = QKVN / 32;           // 48
    const int tm = wid / tilesN;            // 0..255
    const int tn = wid % tilesN;            // 0..47

    const bf16_t* A0 = Xb + (long)(tm * 32 + lq) * DIN;
    const bf16_t* A1 = A0 + 16 * DIN;
    const bf16_t* B0 = Wb + (long)(tn * 32 + lq) * DIN;    // W[n,k], K-contiguous
    const bf16_t* B1 = B0 + 16 * DIN;

    v8f c00 = {}, c01 = {}, c10 = {}, c11 = {};
#pragma unroll 2
    for (int kb = 0; kb < DIN; kb += 32) {
        Frag a0 = load_frag(A0 + kb, half);
        Frag a1 = load_frag(A1 + kb, half);
        Frag b0 = load_frag(B0 + kb, half);
        Frag b1 = load_frag(B1 + kb, half);
        c00 = wmma_bf16(a0, b0, c00);
        c01 = wmma_bf16(a0, b1, c01);
        c10 = wmma_bf16(a1, b0, c10);
        c11 = wmma_bf16(a1, b1, c11);
    }

    const v8f accs[2][2] = { { c00, c01 }, { c10, c11 } };
#pragma unroll
    for (int ni = 0; ni < 2; ++ni) {
        const int n    = tn * 32 + ni * 16 + lq;      // 0..1535
        const float bv = bias[n];
        const int h    = n / (3 * HDIM);              // head
        const int part = (n % (3 * HDIM)) / HDIM;     // 0=q 1=k 2=v
        const int d    = n % HDIM;
#pragma unroll
        for (int mi = 0; mi < 2; ++mi) {
            const v8f acc = accs[mi][ni];
#pragma unroll
            for (int r = 0; r < 8; ++r) {
                const int m  = tm * 32 + mi * 16 + r + half * 8;  // global row
                const int b_ = m >> 12;
                const int s  = m & 4095;
                const float val = acc[r] + bv;
                const long bh = (long)(b_ * NHEAD + h);
                if (part == 0)
                    Qb[(bh * SEQ + s) * HDIM + d] = (bf16_t)(val * 0.125f); // fold 1/sqrt(d)
                else if (part == 1)
                    Kb[(bh * SEQ + s) * HDIM + d] = (bf16_t)val;
                else
                    Vt[(bh * HDIM + d) * SEQ + s] = (bf16_t)val;            // transposed
            }
        }
    }
}

// ---------------- stage 2: flash attention ----------------
// Block = 8 waves, all on one (b,h), consecutive 16-row query tiles.
// 32-key tiles of K and Vt are staged in LDS via async copies, double-buffered.
__global__ __launch_bounds__(256)
void flash_attn_kernel(const bf16_t* __restrict__ Qb,
                       const bf16_t* __restrict__ Kb,
                       const bf16_t* __restrict__ Vt,
                       bf16_t* __restrict__ Ob) {
    __shared__ __align__(16) bf16_t Kl[2][32][HDIM];   // [buf][key][d]   8 KB
    __shared__ __align__(16) bf16_t Vl[2][HDIM][32];   // [buf][d][key]   8 KB
    __shared__ bf16_t Pl[8][16][32];                   // P relayout      8 KB

    const int tid  = threadIdx.x;
    const int wv   = tid >> 5;
    const int lane = tid & 31;
    const int half = lane >> 4;
    const int lq   = lane & 15;

    const int bh = blockIdx.x >> 5;                 // 0..15
    const int qt = (blockIdx.x & 31) * 8 + wv;      // 0..255

    const bf16_t* Qt = Qb + ((long)bh * SEQ + qt * 16) * HDIM;
    const bf16_t* Kh = Kb + (long)bh * SEQ * HDIM;
    const bf16_t* Vh = Vt + (long)bh * HDIM * SEQ;

    // Q fragments (16 rows x 64 d) as two K=32 chunks; A row = lane%16
    const Frag qa0 = load_frag(Qt + (long)lq * HDIM + 0,  half);
    const Frag qa1 = load_frag(Qt + (long)lq * HDIM + 32, half);

    // per-buffer LDS byte offsets (uniform across block)
    const unsigned kOff[2] = { lds_off(&Kl[0][0][0]), lds_off(&Kl[1][0][0]) };
    const unsigned vOff[2] = { lds_off(&Vl[0][0][0]), lds_off(&Vl[1][0][0]) };

    // per-thread chunk assignment for tile staging (8 KB per tile, 256 threads)
    // K tile: contiguous 4 KB -> thread t copies bytes [t*16, t*16+16)
    // V tile: 64 rows (d) x 64 B -> thread t copies row t/4, 16B-chunk t%4
    const int vd = tid >> 2;                 // 0..63
    const int vc = tid & 3;                  // 0..3

    // prologue: stage tile 0 into buffer 0
    {
        async_b128(kOff[0] + tid * 16, (const char*)(Kh) + tid * 16);
        async_b128(vOff[0] + (unsigned)(vd * 64 + vc * 16),
                   (const char*)(Vh + (long)vd * SEQ + vc * 8));
    }

    v8f o0 = {}, o1 = {}, o2 = {}, o3 = {};
    float mrun[8], lrun[8];
#pragma unroll
    for (int r = 0; r < 8; ++r) { mrun[r] = -1e30f; lrun[r] = 0.0f; }

    const int NTILE = SEQ / 32;              // 128

    for (int i = 0; i < NTILE; ++i) {
        const int buf = i & 1;

        // prefetch next tile (wraps to tile 0 on last iter; keeps count constant)
        {
            const int nkt = ((i + 1) & (NTILE - 1)) * 32;
            const int nb  = buf ^ 1;
            async_b128(kOff[nb] + tid * 16,
                       (const char*)(Kh + (long)nkt * HDIM) + tid * 16);
            async_b128(vOff[nb] + (unsigned)(vd * 64 + vc * 16),
                       (const char*)(Vh + (long)vd * SEQ + nkt + vc * 8));
        }

        // current tile resident when only the 2 prefetch ops remain outstanding
        asm volatile("s_wait_asynccnt 0x2" ::: "memory");
        __syncthreads();

        // ---- logits: S = Q K^T for 32 keys (two 16-wide column groups) ----
        Frag kb0 = load_frag(&Kl[buf][lq][0],       half);
        Frag kb1 = load_frag(&Kl[buf][lq][32],      half);
        v8f s1 = {};
        s1 = wmma_bf16(qa0, kb0, s1);
        s1 = wmma_bf16(qa1, kb1, s1);

        Frag kb2 = load_frag(&Kl[buf][16 + lq][0],  half);
        Frag kb3 = load_frag(&Kl[buf][16 + lq][32], half);
        v8f s2 = {};
        s2 = wmma_bf16(qa0, kb2, s2);
        s2 = wmma_bf16(qa1, kb3, s2);

        // ---- online softmax over the 32 new columns (Q pre-scaled) ----
        float p1[8], p2[8];
#pragma unroll
        for (int r = 0; r < 8; ++r) {
            const float a = s1[r];
            const float b = s2[r];
            float t = fmaxf(a, b);
#pragma unroll
            for (int off = 1; off < 16; off <<= 1)
                t = fmaxf(t, __shfl_xor(t, off, 32));
            const float mn = fmaxf(mrun[r], t);
            const float al = __expf(mrun[r] - mn);
            p1[r] = __expf(a - mn);
            p2[r] = __expf(b - mn);
            float rs = p1[r] + p2[r];
#pragma unroll
            for (int off = 1; off < 16; off <<= 1)
                rs += __shfl_xor(rs, off, 32);
            lrun[r] = lrun[r] * al + rs;
            mrun[r] = mn;
            o0[r] *= al; o1[r] *= al; o2[r] *= al; o3[r] *= al;
        }

        // ---- relayout P (C-layout) -> A-fragment layout via LDS ----
#pragma unroll
        for (int r = 0; r < 8; ++r) {
            const int row = r + half * 8;
            Pl[wv][row][lq]      = (bf16_t)p1[r];
            Pl[wv][row][lq + 16] = (bf16_t)p2[r];
        }
        asm volatile("s_wait_dscnt 0" ::: "memory");
        const Frag pa = load_frag(&Pl[wv][lq][0], half);

        // ---- O += P * V  (Vl rows are d, K-contiguous over the 32 keys) ----
        {
            Frag vb = load_frag(&Vl[buf][0 * 16 + lq][0], half);
            o0 = wmma_bf16(pa, vb, o0);
        }
        {
            Frag vb = load_frag(&Vl[buf][1 * 16 + lq][0], half);
            o1 = wmma_bf16(pa, vb, o1);
        }
        {
            Frag vb = load_frag(&Vl[buf][2 * 16 + lq][0], half);
            o2 = wmma_bf16(pa, vb, o2);
        }
        {
            Frag vb = load_frag(&Vl[buf][3 * 16 + lq][0], half);
            o3 = wmma_bf16(pa, vb, o3);
        }

        // protect the buffer that next iteration's prefetch will overwrite
        __syncthreads();
    }

    // ---- normalize and write out as [B, S, EMB] bf16 ----
    const int b_ = bh >> 3;
    const int h  = bh & 7;
#pragma unroll
    for (int r = 0; r < 8; ++r) {
        const float inv = 1.0f / lrun[r];
        const int row = r + half * 8;
        const int s   = qt * 16 + row;
        const long base = ((long)(b_ * SEQ + s)) * EMB + h * HDIM;
        Ob[base + 0 * 16 + lq] = (bf16_t)(o0[r] * inv);
        Ob[base + 1 * 16 + lq] = (bf16_t)(o1[r] * inv);
        Ob[base + 2 * 16 + lq] = (bf16_t)(o2[r] * inv);
        Ob[base + 3 * 16 + lq] = (bf16_t)(o3[r] * inv);
    }
}

// ---------------- stage 3: output projection ----------------
// out[MTOT x 512] = Ob[MTOT x 512] * Wo^T + b_o, 32x32 tile per wave (fp32 out)
__global__ __launch_bounds__(256)
void out_proj_kernel(const bf16_t* __restrict__ Ob,
                     const bf16_t* __restrict__ Wob,
                     const float*  __restrict__ bias,
                     float* __restrict__ out) {
    const int wid  = (blockIdx.x * blockDim.x + threadIdx.x) >> 5;
    const int lane = threadIdx.x & 31;
    const int half = lane >> 4;
    const int lq   = lane & 15;

    const int tilesN = EMB / 32;            // 16
    const int tm = wid / tilesN;            // 0..255
    const int tn = wid % tilesN;            // 0..15

    const bf16_t* A0 = Ob  + (long)(tm * 32 + lq) * EMB;
    const bf16_t* A1 = A0 + 16 * EMB;
    const bf16_t* B0 = Wob + (long)(tn * 32 + lq) * EMB;
    const bf16_t* B1 = B0 + 16 * EMB;

    v8f c00 = {}, c01 = {}, c10 = {}, c11 = {};
#pragma unroll 2
    for (int kb = 0; kb < EMB; kb += 32) {
        Frag a0 = load_frag(A0 + kb, half);
        Frag a1 = load_frag(A1 + kb, half);
        Frag b0 = load_frag(B0 + kb, half);
        Frag b1 = load_frag(B1 + kb, half);
        c00 = wmma_bf16(a0, b0, c00);
        c01 = wmma_bf16(a0, b1, c01);
        c10 = wmma_bf16(a1, b0, c10);
        c11 = wmma_bf16(a1, b1, c11);
    }

    const v8f accs[2][2] = { { c00, c01 }, { c10, c11 } };
#pragma unroll
    for (int ni = 0; ni < 2; ++ni) {
        const int n    = tn * 32 + ni * 16 + lq;
        const float bv = bias[n];
#pragma unroll
        for (int mi = 0; mi < 2; ++mi) {
            const v8f acc = accs[mi][ni];
#pragma unroll
            for (int r = 0; r < 8; ++r) {
                const int m = tm * 32 + mi * 16 + r + half * 8;
                out[(long)m * EMB + n] = acc[r] + bv;
            }
        }
    }
}

// ---------------- launch ----------------
extern "C" void kernel_launch(void* const* d_in, const int* in_sizes, int n_in,
                              void* d_out, int out_size, void* d_ws, size_t ws_size,
                              hipStream_t stream) {
    const float* X    = (const float*)d_in[0];   // [2,4096,512]
    const float* Wqkv = (const float*)d_in[1];   // [1536,512]
    const float* bqkv = (const float*)d_in[2];   // [1536]
    const float* Wo   = (const float*)d_in[3];   // [512,512]
    const float* bo   = (const float*)d_in[4];   // [512]
    float* out = (float*)d_out;                  // [2,4096,512] fp32

    // workspace carve-up (bf16 buffers)
    char* ws = (char*)d_ws;
    bf16_t* Xb    = (bf16_t*)ws; ws += (size_t)MTOT * DIN * 2;          // 8 MB
    bf16_t* Wqkvb = (bf16_t*)ws; ws += (size_t)QKVN * DIN * 2;          // 1.5 MB
    bf16_t* Wob   = (bf16_t*)ws; ws += (size_t)EMB * EMB * 2;           // 0.5 MB
    bf16_t* Qb    = (bf16_t*)ws; ws += (size_t)BATCH * NHEAD * SEQ * HDIM * 2; // 8 MB
    bf16_t* Kb    = (bf16_t*)ws; ws += (size_t)BATCH * NHEAD * SEQ * HDIM * 2; // 8 MB
    bf16_t* Vt    = (bf16_t*)ws; ws += (size_t)BATCH * NHEAD * HDIM * SEQ * 2; // 8 MB
    bf16_t* Ob    = (bf16_t*)ws; ws += (size_t)MTOT * EMB * 2;          // 8 MB

    // stage 0: converts
    {
        int n = MTOT * DIN;
        cvt_bf16_kernel<<<(n + 255) / 256, 256, 0, stream>>>(X, Xb, n);
    }
    {
        int n = QKVN * DIN;
        cvt_bf16_kernel<<<(n + 255) / 256, 256, 0, stream>>>(Wqkv, Wqkvb, n);
    }
    {
        int n = EMB * EMB;
        cvt_bf16_kernel<<<(n + 255) / 256, 256, 0, stream>>>(Wo, Wob, n);
    }

    // stage 1: QKV projection — (8192/32)*(1536/32) = 12288 waves, 8 per block
    qkv_gemm_kernel<<<12288 / 8, 256, 0, stream>>>(Xb, Wqkvb, bqkv, Qb, Kb, Vt);

    // stage 2: flash attention — 16 bh * 32 blocks, 8 q-tiles per block
    flash_attn_kernel<<<512, 256, 0, stream>>>(Qb, Kb, Vt, Ob);

    // stage 3: output projection — (8192/32)*(512/32) = 4096 waves
    out_proj_kernel<<<4096 / 8, 256, 0, stream>>>(Ob, Wob, bo, out);
}
